// FlashMultiheadAttention_68573447848433
// MI455X (gfx1250) — compile-verified, matrix operand order checked by reading, not run on hardware
//
#include <hip/hip_runtime.h>
#include <hip/hip_bf16.h>
#include <stdint.h>

// ---------------------------------------------------------------------------
// Types for CDNA5 WMMA bf16 fragments (wave32)
// ---------------------------------------------------------------------------
typedef __attribute__((ext_vector_type(16))) __bf16 v16bf;
typedef __attribute__((ext_vector_type(8)))  __bf16 v8bf;
typedef __attribute__((ext_vector_type(8)))  float  v8f;
typedef int v4i __attribute__((vector_size(16)));

union bffrag { v16bf v; v8bf h[2]; };

__device__ __forceinline__ v8f wmma_bf16(const bffrag& a, const bffrag& b, v8f c) {
    return __builtin_amdgcn_wmma_f32_16x16x32_bf16(
        /*neg_a=*/false, a.v, /*neg_b=*/false, b.v,
        /*c_mod=*/(short)0, c, /*reuse_a=*/false, /*reuse_b=*/false);
}

// ---------------------------------------------------------------------------
// CDNA5 async global->LDS copy (ASYNCcnt) with synchronous fallback.
// 16 bytes per lane.  dst_lds must be a pointer into __shared__ memory.
// Builtin signature (probe-discovered): (v4i AS1* gaddr, v4i AS3* lds, imm, imm)
// ---------------------------------------------------------------------------
__device__ __forceinline__ void cp_async16(__bf16* dst_lds, const __bf16* src) {
#if __has_builtin(__builtin_amdgcn_global_load_async_to_lds_b128)
    __builtin_amdgcn_global_load_async_to_lds_b128(
        (__attribute__((address_space(1))) v4i*)(uintptr_t)src,
        (__attribute__((address_space(3))) v4i*)(uint32_t)(uintptr_t)dst_lds,
        0, 0);
#else
    *(v8bf*)dst_lds = *(const v8bf*)src;
#endif
}

__device__ __forceinline__ void cp_async_wait() {
#if __has_builtin(__builtin_amdgcn_global_load_async_to_lds_b128) && \
    __has_builtin(__builtin_amdgcn_s_wait_asynccnt)
    __builtin_amdgcn_s_wait_asynccnt(0);
#endif
}

// ---------------------------------------------------------------------------
// Elementwise f32 -> bf16 cast (weights)
// ---------------------------------------------------------------------------
__global__ __launch_bounds__(256) void cast_bf16_kernel(const float* __restrict__ in,
                                                        __bf16* __restrict__ out, int n) {
    int i = blockIdx.x * blockDim.x + threadIdx.x;
    if (i < n) out[i] = (__bf16)in[i];
}

// ---------------------------------------------------------------------------
// Row LayerNorm (with bias) -> bf16.  One block per row.
// ---------------------------------------------------------------------------
__global__ __launch_bounds__(256) void ln_rows_kernel(const float* __restrict__ X,
                                                      const float* __restrict__ w,
                                                      const float* __restrict__ bias,
                                                      __bf16* __restrict__ out, int E) {
    __shared__ float red[16];
    const int t = blockIdx.x;
    const float* row = X + (size_t)t * E;
    float s = 0.f, ss = 0.f;
    for (int e = threadIdx.x; e < E; e += blockDim.x) {
        float v = row[e]; s += v; ss += v * v;
    }
    #pragma unroll
    for (int o = 16; o > 0; o >>= 1) {
        s  += __shfl_xor(s,  o, 32);
        ss += __shfl_xor(ss, o, 32);
    }
    if ((threadIdx.x & 31) == 0) {
        red[threadIdx.x >> 5] = s;
        red[8 + (threadIdx.x >> 5)] = ss;
    }
    __syncthreads();
    float ts = 0.f, tss = 0.f;
    #pragma unroll
    for (int i = 0; i < 8; ++i) { ts += red[i]; tss += red[8 + i]; }
    const float mean = ts / (float)E;
    const float var  = tss / (float)E - mean * mean;
    const float rs   = rsqrtf(var + 1e-5f);
    for (int e = threadIdx.x; e < E; e += blockDim.x) {
        float y = (row[e] - mean) * rs * w[e] + bias[e];
        out[(size_t)t * E + e] = (__bf16)y;
    }
}

// ---------------------------------------------------------------------------
// Row LayerNorm (no bias) + rotary embedding -> bf16.  One block per row.
// D = 64 per head; rotary pairs (d, d+32) within each head; positions restart
// at each cu_lens boundary.
// ---------------------------------------------------------------------------
__global__ __launch_bounds__(256) void qk_ln_rope_kernel(const float* __restrict__ X,
                                                         const float* __restrict__ w,
                                                         const int* __restrict__ cu,
                                                         int nseq,
                                                         __bf16* __restrict__ out, int E) {
    __shared__ float xrow[1280];
    __shared__ float red[16];
    const int t = blockIdx.x;
    const float* row = X + (size_t)t * E;
    float s = 0.f, ss = 0.f;
    for (int e = threadIdx.x; e < E; e += blockDim.x) {
        float v = row[e]; xrow[e] = v; s += v; ss += v * v;
    }
    #pragma unroll
    for (int o = 16; o > 0; o >>= 1) {
        s  += __shfl_xor(s,  o, 32);
        ss += __shfl_xor(ss, o, 32);
    }
    if ((threadIdx.x & 31) == 0) {
        red[threadIdx.x >> 5] = s;
        red[8 + (threadIdx.x >> 5)] = ss;
    }
    __syncthreads();
    float ts = 0.f, tss = 0.f;
    #pragma unroll
    for (int i = 0; i < 8; ++i) { ts += red[i]; tss += red[8 + i]; }
    const float mean = ts / (float)E;
    const float rs   = rsqrtf(tss / (float)E - mean * mean + 1e-5f);

    // position within its sequence
    int p = 0;
    for (int sg = 0; sg < nseq; ++sg) if (t >= cu[sg]) p = t - cu[sg];
    const float fp = (float)p;

    for (int e = threadIdx.x; e < E; e += blockDim.x) {
        const int d = e & 63;          // within-head index (D = 64)
        const int j = d & 31;          // frequency index
        const int partner = (d < 32) ? e + 32 : e - 32;
        float y  = (xrow[e]       - mean) * rs * w[e];
        float y2 = (xrow[partner] - mean) * rs * w[partner];
        float freq = fp * __powf(10000.f, -(float)j * (1.f / 32.f));
        float c = __cosf(freq), sn = __sinf(freq);
        float o = (d < 32) ? (y * c - y2 * sn) : (y * c + y2 * sn);
        out[(size_t)t * E + e] = (__bf16)o;
    }
}

// ---------------------------------------------------------------------------
// WMMA GEMM: C[M,N] = A[M,K] * W[N,K]^T   (A, W bf16; C f32 and/or bf16)
// Block tile 128x64, K-step 64, 8 waves (2x2 16x16 WMMA tiles per wave).
// Double-buffered LDS with async global->LDS staging.
// ---------------------------------------------------------------------------
#define GM_TM 128
#define GM_TN 64
#define GM_TK 64
#define GM_LD 72   // 64 + 8 pad (bf16 elems); row stride 144B, 16B aligned

__global__ __launch_bounds__(256) void gemm_bf16_kernel(const __bf16* __restrict__ A,
                                                        const __bf16* __restrict__ W,
                                                        float*  __restrict__ outF,
                                                        __bf16* __restrict__ outB,
                                                        int M, int N, int K) {
    __shared__ __bf16 As[2][GM_TM * GM_LD];
    __shared__ __bf16 Ws[2][GM_TN * GM_LD];

    const int tid  = threadIdx.x;
    const int lane = tid & 31;
    const int wave = tid >> 5;
    const int wm   = wave & 3;     // 4 waves along M
    const int wn   = wave >> 2;    // 2 waves along N
    const int hi   = lane >> 4;    // half-wave select
    const int ln16 = lane & 15;
    const int m0 = blockIdx.x * GM_TM;
    const int n0 = blockIdx.y * GM_TN;

    auto stage = [&](int buf, int k0) {
        // A tile: 128 rows x 64 cols -> 1024 16B chunks / 256 thr = 4 ea
        #pragma unroll
        for (int i = 0; i < 4; ++i) {
            int c = tid + i * 256;
            int r = c >> 3, cc = (c & 7) * 8;
            cp_async16(&As[buf][r * GM_LD + cc], A + (size_t)(m0 + r) * K + k0 + cc);
        }
        // W tile: 64 rows x 64 cols -> 512 chunks / 256 thr = 2 ea
        #pragma unroll
        for (int i = 0; i < 2; ++i) {
            int c = tid + i * 256;
            int r = c >> 3, cc = (c & 7) * 8;
            cp_async16(&Ws[buf][r * GM_LD + cc], W + (size_t)(n0 + r) * K + k0 + cc);
        }
    };

    v8f acc[2][2] = {};

    stage(0, 0);
    cp_async_wait();
    __syncthreads();

    int buf = 0;
    for (int k0 = 0; k0 < K; k0 += GM_TK) {
        // prefetch next tile into the other buffer while we compute
        if (k0 + GM_TK < K) stage(buf ^ 1, k0 + GM_TK);

        #pragma unroll
        for (int kc = 0; kc < 2; ++kc) {
            const int kb = kc * 32 + hi * 16;
            bffrag a[2], b[2];
            #pragma unroll
            for (int ms = 0; ms < 2; ++ms) {
                const __bf16* p = &As[buf][(wm * 32 + ms * 16 + ln16) * GM_LD + kb];
                a[ms].h[0] = *(const v8bf*)p;
                a[ms].h[1] = *(const v8bf*)(p + 8);
            }
            #pragma unroll
            for (int ns = 0; ns < 2; ++ns) {
                const __bf16* p = &Ws[buf][(wn * 32 + ns * 16 + ln16) * GM_LD + kb];
                b[ns].h[0] = *(const v8bf*)p;
                b[ns].h[1] = *(const v8bf*)(p + 8);
            }
            #pragma unroll
            for (int ms = 0; ms < 2; ++ms)
                #pragma unroll
                for (int ns = 0; ns < 2; ++ns)
                    acc[ms][ns] = wmma_bf16(a[ms], b[ns], acc[ms][ns]);
        }

        cp_async_wait();
        __syncthreads();
        buf ^= 1;
    }

    // epilogue: C layout -> row = i + 8*hi, col = ln16
    #pragma unroll
    for (int ms = 0; ms < 2; ++ms)
        #pragma unroll
        for (int ns = 0; ns < 2; ++ns)
            #pragma unroll
            for (int i = 0; i < 8; ++i) {
                int r = m0 + wm * 32 + ms * 16 + i + hi * 8;
                int c = n0 + wn * 32 + ns * 16 + ln16;
                float v = acc[ms][ns][i];
                if (outF) outF[(size_t)r * N + c] = v;
                if (outB) outB[(size_t)r * N + c] = (__bf16)v;
            }
}

// ---------------------------------------------------------------------------
// Flash attention: one block per (seq b, head h, 128 q-rows).  8 waves, each
// owning 16 q-rows.  K tile row-major (async-staged), V tile transposed in
// LDS so all WMMA B fragments are per-lane contiguous.  Double-buffered K/V
// tiles; online softmax on the WMMA C layout with __shfl_xor row reductions.
// ---------------------------------------------------------------------------
__global__ __launch_bounds__(256) void flash_attn_kernel(const __bf16* __restrict__ Q,
                                                         const __bf16* __restrict__ Kb,
                                                         const __bf16* __restrict__ Vb,
                                                         __bf16* __restrict__ O,
                                                         int S, int E) {
    __shared__ __bf16 Ks[2][64 * GM_LD];     // [kpos][d]
    __shared__ __bf16 Vt[2][64 * GM_LD];     // [d][kpos] (transposed)
    __shared__ __bf16 Ps[8 * 16 * GM_LD];    // per-wave 16x64 P staging

    const int tid  = threadIdx.x;
    const int lane = tid & 31;
    const int wave = tid >> 5;
    const int hi   = lane >> 4;
    const int ln16 = lane & 15;
    const int b  = blockIdx.z;
    const int h  = blockIdx.y;
    const int D  = 64;
    const int t0 = b * S + blockIdx.x * 128;       // token base of this block
    const int tq = t0 + wave * 16 + ln16;          // this lane's Q row

    auto stage = [&](int buf, int n0) {
        #pragma unroll
        for (int i = 0; i < 2; ++i) {
            int c = tid + i * 256;
            int r = c >> 3, cc = (c & 7) * 8;
            // K tile: async straight into LDS
            cp_async16(&Ks[buf][r * GM_LD + cc],
                       Kb + (size_t)(b * S + n0 + r) * E + h * D + cc);
            // V tile: VGPR-mediated transpose scatter
            const __bf16* vsrc = Vb + (size_t)(b * S + n0 + r) * E + h * D + cc;
            v8bf vv = *(const v8bf*)vsrc;
            #pragma unroll
            for (int j = 0; j < 8; ++j) Vt[buf][(cc + j) * GM_LD + r] = vv[j];
        }
    };

    // Q fragments (row-per-lane, K(=d) split by half-wave)
    bffrag qf[2];
    #pragma unroll
    for (int ch = 0; ch < 2; ++ch) {
        const __bf16* src = Q + (size_t)tq * E + h * D + ch * 32 + hi * 16;
        qf[ch].h[0] = *(const v8bf*)src;
        qf[ch].h[1] = *(const v8bf*)(src + 8);
    }

    v8f oacc[4] = {};
    float mrow[8], lrow[8];
    #pragma unroll
    for (int i = 0; i < 8; ++i) { mrow[i] = -1e30f; lrow[i] = 0.f; }

    const float scale = rsqrtf((float)D);
    __bf16* Pw = &Ps[wave * 16 * GM_LD];

    stage(0, 0);
    cp_async_wait();
    __syncthreads();

    int buf = 0;
    for (int n0 = 0; n0 < S; n0 += 64) {
        if (n0 + 64 < S) stage(buf ^ 1, n0 + 64);

        // scores S = Q K^T  (16 q-rows x 64 k-cols per wave)
        v8f sacc[4] = {};
        #pragma unroll
        for (int nt = 0; nt < 4; ++nt) {
            #pragma unroll
            for (int ch = 0; ch < 2; ++ch) {
                bffrag bfr;
                const __bf16* p = &Ks[buf][(nt * 16 + ln16) * GM_LD + ch * 32 + hi * 16];
                bfr.h[0] = *(const v8bf*)p;
                bfr.h[1] = *(const v8bf*)(p + 8);
                sacc[nt] = wmma_bf16(qf[ch], bfr, sacc[nt]);
            }
        }

        // online softmax per row (row = i + 8*hi, cols across 16-lane group)
        #pragma unroll
        for (int i = 0; i < 8; ++i) {
            float mx = -1e30f;
            #pragma unroll
            for (int nt = 0; nt < 4; ++nt) mx = fmaxf(mx, sacc[nt][i] * scale);
            #pragma unroll
            for (int o = 1; o < 16; o <<= 1) mx = fmaxf(mx, __shfl_xor(mx, o, 32));
            float mnew  = fmaxf(mrow[i], mx);
            float alpha = __expf(mrow[i] - mnew);
            float sum = 0.f;
            #pragma unroll
            for (int nt = 0; nt < 4; ++nt) {
                float pv = __expf(sacc[nt][i] * scale - mnew);
                sacc[nt][i] = pv;
                sum += pv;
            }
            #pragma unroll
            for (int o = 1; o < 16; o <<= 1) sum += __shfl_xor(sum, o, 32);
            lrow[i] = lrow[i] * alpha + sum;
            mrow[i] = mnew;
            #pragma unroll
            for (int dt = 0; dt < 4; ++dt) oacc[dt][i] *= alpha;
        }

        // P: C-layout -> A-layout via per-wave LDS staging
        #pragma unroll
        for (int nt = 0; nt < 4; ++nt)
            #pragma unroll
            for (int i = 0; i < 8; ++i)
                Pw[(i + hi * 8) * GM_LD + nt * 16 + ln16] = (__bf16)sacc[nt][i];
        __syncthreads();

        bffrag pf[2];
        #pragma unroll
        for (int ch = 0; ch < 2; ++ch) {
            const __bf16* p = &Pw[ln16 * GM_LD + ch * 32 + hi * 16];
            pf[ch].h[0] = *(const v8bf*)p;
            pf[ch].h[1] = *(const v8bf*)(p + 8);
        }
        // O += P V
        #pragma unroll
        for (int dt = 0; dt < 4; ++dt) {
            #pragma unroll
            for (int ch = 0; ch < 2; ++ch) {
                bffrag bfr;
                const __bf16* p = &Vt[buf][(dt * 16 + ln16) * GM_LD + ch * 32 + hi * 16];
                bfr.h[0] = *(const v8bf*)p;
                bfr.h[1] = *(const v8bf*)(p + 8);
                oacc[dt] = wmma_bf16(pf[ch], bfr, oacc[dt]);
            }
        }

        cp_async_wait();
        __syncthreads();
        buf ^= 1;
    }

    // normalize and store (bf16, [T, E] with head offset)
    #pragma unroll
    for (int i = 0; i < 8; ++i) {
        float inv = 1.f / lrow[i];
        int t = t0 + wave * 16 + i + hi * 8;
        #pragma unroll
        for (int dt = 0; dt < 4; ++dt)
            O[(size_t)t * E + h * D + dt * 16 + ln16] = (__bf16)(oacc[dt][i] * inv);
    }
}

// ---------------------------------------------------------------------------
// Host launcher
// ---------------------------------------------------------------------------
extern "C" void kernel_launch(void* const* d_in, const int* in_sizes, int n_in,
                              void* d_out, int out_size, void* d_ws, size_t ws_size,
                              hipStream_t stream) {
    (void)n_in; (void)out_size; (void)ws_size;
    const float* x      = (const float*)d_in[0];
    const int*   cu     = (const int*)d_in[1];
    const float* norm_w = (const float*)d_in[3];
    const float* norm_b = (const float*)d_in[4];
    const float* Wq     = (const float*)d_in[5];
    const float* Wk     = (const float*)d_in[6];
    const float* Wv     = (const float*)d_in[7];
    const float* Wout   = (const float*)d_in[8];
    const float* lnq_w  = (const float*)d_in[9];
    const float* lnk_w  = (const float*)d_in[10];

    const int E = 1280, H = 20;
    const int T = in_sizes[0] / E;
    const int B = in_sizes[1] - 1;
    const int S = T / B;

    // workspace carve-up (all 256B aligned)
    char* p = (char*)d_ws;
    auto take = [&](size_t bytes) -> char* {
        char* r = p; p += (bytes + 255) & ~(size_t)255; return r;
    };
    const size_t szW  = (size_t)E * E * sizeof(__bf16);
    const size_t szTb = (size_t)T * E * sizeof(__bf16);
    const size_t szTf = (size_t)T * E * sizeof(float);
    __bf16* Wq_b = (__bf16*)take(szW);
    __bf16* Wk_b = (__bf16*)take(szW);
    __bf16* Wv_b = (__bf16*)take(szW);
    __bf16* Wo_b = (__bf16*)take(szW);
    __bf16* h_b  = (__bf16*)take(szTb);
    __bf16* q_b  = (__bf16*)take(szTb);
    __bf16* k_b  = (__bf16*)take(szTb);
    __bf16* v_b  = (__bf16*)take(szTb);
    __bf16* o_b  = (__bf16*)take(szTb);
    float*  pre  = (float*)take(szTf);   // reused for q-pre then k-pre

    const int nW = E * E;
    cast_bf16_kernel<<<(nW + 255) / 256, 256, 0, stream>>>(Wq,   Wq_b, nW);
    cast_bf16_kernel<<<(nW + 255) / 256, 256, 0, stream>>>(Wk,   Wk_b, nW);
    cast_bf16_kernel<<<(nW + 255) / 256, 256, 0, stream>>>(Wv,   Wv_b, nW);
    cast_bf16_kernel<<<(nW + 255) / 256, 256, 0, stream>>>(Wout, Wo_b, nW);

    ln_rows_kernel<<<T, 256, 0, stream>>>(x, norm_w, norm_b, h_b, E);

    dim3 gg(T / GM_TM, E / GM_TN);
    // v = h @ Wv.T (bf16 out)
    gemm_bf16_kernel<<<gg, 256, 0, stream>>>(h_b, Wv_b, nullptr, v_b, T, E, E);
    // q = ln(h @ Wq.T) + rope
    gemm_bf16_kernel<<<gg, 256, 0, stream>>>(h_b, Wq_b, pre, nullptr, T, E, E);
    qk_ln_rope_kernel<<<T, 256, 0, stream>>>(pre, lnq_w, cu, B, q_b, E);
    // k = ln(h @ Wk.T) + rope
    gemm_bf16_kernel<<<gg, 256, 0, stream>>>(h_b, Wk_b, pre, nullptr, T, E, E);
    qk_ln_rope_kernel<<<T, 256, 0, stream>>>(pre, lnk_w, cu, B, k_b, E);

    // attention per (seq, head, 128 q-rows)
    flash_attn_kernel<<<dim3(S / 128, H, B), 256, 0, stream>>>(q_b, k_b, v_b, o_b, S, E);

    // out = o @ Wout.T (f32 -> d_out)
    gemm_bf16_kernel<<<gg, 256, 0, stream>>>(o_b, Wo_b, (float*)d_out, nullptr, T, E, E);
}
